// PerformerAttention_5970004541506
// MI455X (gfx1250) — compile-verified, matrix operand order checked by reading, not run on hardware
//
#include <hip/hip_runtime.h>

// ---------------------------------------------------------------------------
// Performer (FAVOR+) attention for MI455X (gfx1250), bf16 WMMA, f32 accum.
// B=4,H=16 -> 64 heads; N=4096; D=64; M=266 (padded to 288 = 9 * 32).
// Phase 2 double-buffers its k/v chunk staging with CDNA5 async global->LDS
// copies (ASYNCcnt / s_wait_asynccnt); all WMMA fragments are contiguous 16B
// LDS loads; f32->bf16 conversion happens once per element via packed stores.
// ---------------------------------------------------------------------------

typedef __attribute__((ext_vector_type(16))) __bf16 v16bf;
typedef __attribute__((ext_vector_type(8)))  __bf16 v8bf;
typedef __attribute__((ext_vector_type(8)))  float  v8f;

#define NHEAD 64
#define NSEQ  4096
#define DDIM  64
#define MFEAT 266
#define MPAD  288                 // 18 tiles of 16, 9 K-steps of 32
#define RATIO 0.06131393f         // 266^-0.5
#define CNORM 0.35355339f         // 64^-0.25
#define DIAGC 0.0625f             // 0.5 * 64^-0.5
#define EPSF  1e-4f
#define NEGINF (-3.4e38f)

__device__ __forceinline__ unsigned bfbits(float f) {
  unsigned u = __builtin_bit_cast(unsigned, f);
  return (u + 0x7FFFu + ((u >> 16) & 1u)) >> 16;   // round-to-nearest-even
}
__device__ __forceinline__ __bf16 f2bf(float f) {
  return __builtin_bit_cast(__bf16, (unsigned short)bfbits(f));
}
__device__ __forceinline__ void store_bf16x4(__bf16* p, float4 f) {
  uint2 u;
  u.x = bfbits(f.x) | (bfbits(f.y) << 16);
  u.y = bfbits(f.z) | (bfbits(f.w) << 16);
  *reinterpret_cast<uint2*>(p) = u;
}
__device__ __forceinline__ void store_bf16x8(__bf16* p, const v8f& c) {
  uint4 u;
  u.x = bfbits(c[0]) | (bfbits(c[1]) << 16);
  u.y = bfbits(c[2]) | (bfbits(c[3]) << 16);
  u.z = bfbits(c[4]) | (bfbits(c[5]) << 16);
  u.w = bfbits(c[6]) | (bfbits(c[7]) << 16);
  *reinterpret_cast<uint4*>(p) = u;
}

__device__ __forceinline__ v8f wmma_bf16(v16bf a, v16bf b, v8f c) {
  return __builtin_amdgcn_wmma_f32_16x16x32_bf16(false, a, false, b,
                                                 (short)0, c, false, false);
}

// Fragment from a bf16 row in LDS (K-contiguous, 16B-aligned): per CDNA5
// 16-bit A/B layout, lane half selects kbase (caller passes (lane>>4)*8);
// elements 0..7 = K kbase+0..7, elements 8..15 = K kbase+16..23.
__device__ __forceinline__ v16bf frag_bf16_lds(const __bf16* rowbase, int kbase) {
  v8bf lo = __builtin_bit_cast(v8bf, *reinterpret_cast<const uint4*>(rowbase + kbase));
  v8bf hi = __builtin_bit_cast(v8bf, *reinterpret_cast<const uint4*>(rowbase + kbase + 16));
  v16bf f;
#pragma unroll
  for (int j = 0; j < 8; ++j) { f[j] = lo[j]; f[j + 8] = hi[j]; }
  return f;
}

// Fragment straight from an f32 row in global (phase 1 A only; amortized 17x).
__device__ __forceinline__ v16bf frag_f32(const float* rowbase, int kbase) {
  const float4* p0 = reinterpret_cast<const float4*>(rowbase + kbase);
  const float4* p1 = reinterpret_cast<const float4*>(rowbase + kbase + 16);
  float4 a = p0[0], b = p0[1], c = p1[0], d = p1[1];
  v16bf f;
  f[0]=f2bf(a.x); f[1]=f2bf(a.y); f[2]=f2bf(a.z);  f[3]=f2bf(a.w);
  f[4]=f2bf(b.x); f[5]=f2bf(b.y); f[6]=f2bf(b.z);  f[7]=f2bf(b.w);
  f[8]=f2bf(c.x); f[9]=f2bf(c.y); f[10]=f2bf(c.z); f[11]=f2bf(c.w);
  f[12]=f2bf(d.x);f[13]=f2bf(d.y);f[14]=f2bf(d.z); f[15]=f2bf(d.w);
  return f;
}

// Stage proj (scaled by CNORM, rows >= MFEAT zero) into LDS bf16, packed.
__device__ __forceinline__ void stage_proj(const float* __restrict__ proj,
                                           __bf16* projL, int tid) {
  for (int idx4 = tid; idx4 < MPAD * DDIM / 4; idx4 += 256) {
    int m = idx4 >> 4;                       // 16 float4 per 64-wide row
    float4 f = make_float4(0.f, 0.f, 0.f, 0.f);
    if (m < MFEAT) {
      f = *reinterpret_cast<const float4*>(proj + idx4 * 4);
      f.x *= CNORM; f.y *= CNORM; f.z *= CNORM; f.w *= CNORM;
    }
    store_bf16x4(projL + idx4 * 4, f);
  }
}

// Async copy of one contiguous 8 KB chunk (32 rows x 64 f32) global -> LDS.
// Per-lane b128; tracked by ASYNCcnt. LDS address = low 32 bits of the
// generic pointer (ISA flat->LDS aperture mapping). saddr form: 64-bit SGPR
// base + 32-bit per-lane byte offset; inst offset covers the second half.
__device__ __forceinline__ void async_copy_8k(const float* gbase, float* ldsbase,
                                              int tid) {
  unsigned l = (unsigned)(size_t)(void*)ldsbase + (unsigned)(tid * 16);
  unsigned voff = (unsigned)(tid * 16);
  unsigned long long sbase = (unsigned long long)(size_t)gbase;
  asm volatile("global_load_async_to_lds_b128 %0, %1, %2 offset:0"
               :: "v"(l), "v"(voff), "s"(sbase) : "memory");
  asm volatile("global_load_async_to_lds_b128 %0, %1, %2 offset:4096"
               :: "v"(l), "v"(voff), "s"(sbase) : "memory");
}
__device__ __forceinline__ void wait_async0() {
  asm volatile("s_wait_asynccnt 0x0" ::: "memory");
}

// ---------------------------------------------------------------------------
// Phase 1: per-head max of k_dash = (k * CNORM) @ proj^T   (global stabilizer)
// ---------------------------------------------------------------------------
__global__ void __launch_bounds__(256)
perf_kmax(const float* __restrict__ k, const float* __restrict__ proj,
          float* __restrict__ kmax_out) {
  extern __shared__ char smem[];
  __bf16* projL = reinterpret_cast<__bf16*>(smem);                  // MPAD*DDIM
  float*  red   = reinterpret_cast<float*>(smem + MPAD * DDIM * 2); // 8
  const int tid = threadIdx.x, lane = tid & 31, wv = tid >> 5;
  const int l15 = lane & 15, kb = (lane >> 4) * 8;
  const int head = blockIdx.x;
  const float* kh = k + (size_t)head * NSEQ * DDIM;

  stage_proj(proj, projL, tid);
  __syncthreads();

  float mx = NEGINF;
  for (int nt = wv; nt < NSEQ / 16; nt += 8) {
    const float* rowb = kh + (size_t)(nt * 16 + l15) * DDIM;
    v16bf a0 = frag_f32(rowb, kb);
    v16bf a1 = frag_f32(rowb, kb + 32);
    for (int mt = 0; mt < 17; ++mt) {        // covers m in [0,272)
      const __bf16* pb = projL + (mt * 16 + l15) * DDIM;
      v16bf b0 = frag_bf16_lds(pb, kb);
      v16bf b1 = frag_bf16_lds(pb, kb + 32);
      v8f c = {};
      c = wmma_bf16(a0, b0, c);
      c = wmma_bf16(a1, b1, c);
      if (mt * 16 + l15 < MFEAT) {
#pragma unroll
        for (int i = 0; i < 8; ++i) mx = fmaxf(mx, c[i]);
      }
    }
  }
#pragma unroll
  for (int off = 16; off >= 1; off >>= 1)
    mx = fmaxf(mx, __shfl_xor(mx, off, 32));
  if (lane == 0) red[wv] = mx;
  __syncthreads();
  if (tid == 0) {
    float m2 = red[0];
    for (int i = 1; i < 8; ++i) m2 = fmaxf(m2, red[i]);
    kmax_out[head] = m2;
  }
}

// ---------------------------------------------------------------------------
// Phase 2: per head -> k'; context = k'^T @ v (stored [d][m]); k_cumsum.
// Double-buffered async global->LDS staging of k/v chunks.
// ---------------------------------------------------------------------------
__global__ void __launch_bounds__(256)
perf_context(const float* __restrict__ k, const float* __restrict__ v,
             const float* __restrict__ proj, const float* __restrict__ kmax_in,
             float* __restrict__ kcum_out, float* __restrict__ ctx_out) {
  extern __shared__ char smem[];
  __bf16* projL = reinterpret_cast<__bf16*>(smem);   // MPAD*DDIM
  __bf16* kbL   = projL + MPAD * DDIM;               // 32*DDIM  [n][d] bf16
  __bf16* vTL   = kbL + 32 * DDIM;                   // DDIM*32  [d][n] bf16
  __bf16* kpT   = vTL + DDIM * 32;                   // MPAD*32  [m][n] bf16 (k')
  float*  kcumW = reinterpret_cast<float*>(kpT + MPAD * 32); // 8*MPAD
  float*  diagL = kcumW + 8 * MPAD;                  // 32
  float*  kbuf  = diagL + 32;                        // 2 * 32*DDIM f32 (async)
  float*  vbuf  = kbuf + 2 * 32 * DDIM;              // 2 * 32*DDIM f32 (async)

  const int tid = threadIdx.x, lane = tid & 31, wv = tid >> 5;
  const int l15 = lane & 15, kb = (lane >> 4) * 8;
  const int head = blockIdx.x;
  const float* kh = k + (size_t)head * NSEQ * DDIM;
  const float* vh = v + (size_t)head * NSEQ * DDIM;

  // prefetch chunk 0 into buffer parity 0 (async, overlapped with proj stage)
  async_copy_8k(kh, kbuf, tid);
  async_copy_8k(vh, vbuf, tid);

  stage_proj(proj, projL, tid);
  for (int idx = tid; idx < 8 * MPAD; idx += 256) kcumW[idx] = 0.0f;

  float kmax = NEGINF;
  for (int i = 0; i < NHEAD; ++i) kmax = fmaxf(kmax, kmax_in[i]);

  v8f zero = {};
  v8f acc[9];
#pragma unroll
  for (int t = 0; t < 9; ++t) acc[t] = zero;   // wave wv owns tiles wv+8*t of 72

  int p = 0;
  for (int ci = 0; ci < NSEQ / 32; ++ci) {
    wait_async0();                 // this wave's async copies into buf[p] done
    __syncthreads();               // ... and everyone else's too

    if (ci + 1 < NSEQ / 32) {      // prefetch next chunk into the other buffer
      const size_t off = (size_t)(ci + 1) * 32 * DDIM;
      async_copy_8k(kh + off, kbuf + (1 - p) * 32 * DDIM, tid);
      async_copy_8k(vh + off, vbuf + (1 - p) * 32 * DDIM, tid);
    }

    const float* kbp = kbuf + p * 32 * DDIM;
    const float* vbp = vbuf + p * 32 * DDIM;
    // convert: k -> bf16 row-major, v -> bf16 transposed [d][n]
    for (int idx4 = tid; idx4 < 32 * DDIM / 4; idx4 += 256) {
      int n = idx4 >> 4, d0 = (idx4 & 15) * 4;
      float4 fk = reinterpret_cast<const float4*>(kbp)[idx4];
      float4 fv = reinterpret_cast<const float4*>(vbp)[idx4];
      store_bf16x4(kbL + idx4 * 4, fk);
      vTL[(d0 + 0) * 32 + n] = f2bf(fv.x);
      vTL[(d0 + 1) * 32 + n] = f2bf(fv.y);
      vTL[(d0 + 2) * 32 + n] = f2bf(fv.z);
      vTL[(d0 + 3) * 32 + n] = f2bf(fv.w);
    }
    if (tid < 32) {                // diag from the f32 LDS copy (rotated: no
      const float4* kr = reinterpret_cast<const float4*>(kbp) + tid * 16;
      float s = 0.f;               //  bank conflicts)
#pragma unroll
      for (int j = 0; j < 16; ++j) {
        float4 f = kr[(tid + j) & 15];
        s += f.x * f.x + f.y * f.y + f.z * f.z + f.w * f.w;
      }
      diagL[tid] = DIAGC * s;
    }
    __syncthreads();

    // GEMM1: k_dash chunk (2 n-tiles x 18 m-tiles) -> k' transposed in LDS
    for (int t = wv; t < 36; t += 8) {
      int nt = t / 18, mt = t % 18;
      const __bf16* ab = kbL + (nt * 16 + l15) * DDIM;
      v16bf a0 = frag_bf16_lds(ab, kb);
      v16bf a1 = frag_bf16_lds(ab, kb + 32);
      const __bf16* pb = projL + (mt * 16 + l15) * DDIM;
      v16bf b0 = frag_bf16_lds(pb, kb);
      v16bf b1 = frag_bf16_lds(pb, kb + 32);
      v8f c = {};
      c = wmma_bf16(a0, b0, c);
      c = wmma_bf16(a1, b1, c);
      int col = mt * 16 + l15;                 // feature index m (per lane)
      int r0  = nt * 16 + ((lane >> 4) << 3);  // chunk-local n of element 0
      v8f val = zero;
      float psum = 0.f;
      if (col < MFEAT) {
#pragma unroll
        for (int i = 0; i < 8; ++i) {
          float x = RATIO * (__expf(c[i] - diagL[r0 + i] - kmax) + EPSF);
          val[i] = x;
          psum += x;
        }
      }
      store_bf16x8(kpT + col * 32 + r0, val);  // one ds_store_b128
      psum += __shfl_xor(psum, 16, 32);        // combine half-lanes (same col)
      if (lane < 16 && col < MFEAT) kcumW[wv * MPAD + col] += psum;
    }
    __syncthreads();

    // GEMM2: context += k'^T @ v  (72 tiles, K=32; all frags contiguous LDS)
#pragma unroll
    for (int ti = 0; ti < 9; ++ti) {
      int t = wv + ti * 8;
      int mt = t >> 2, dt = t & 3;
      v16bf a = frag_bf16_lds(kpT + (mt * 16 + l15) * 32, kb);
      v16bf b = frag_bf16_lds(vTL + (dt * 16 + l15) * 32, kb);
      acc[ti] = wmma_bf16(a, b, acc[ti]);
    }
    p ^= 1;
    // top-of-loop wait+barrier orders the next convert after these GEMMs
  }
  __syncthreads();

  for (int idx = tid; idx < MPAD; idx += 256) {
    float s = 0.f;
#pragma unroll
    for (int w = 0; w < 8; ++w) s += kcumW[w * MPAD + idx];
    kcum_out[head * MPAD + idx] = s;
  }
  // write context TRANSPOSED: ctx[d][m]; per lane m runs consecutively
  float* ctxh = ctx_out + (size_t)head * DDIM * MPAD;
#pragma unroll
  for (int ti = 0; ti < 9; ++ti) {
    int t = wv + ti * 8;
    int mt = t >> 2, dt = t & 3;
    int d  = dt * 16 + l15;
    int m0 = mt * 16 + ((lane >> 4) << 3);
    float4 lo = make_float4(acc[ti][0], acc[ti][1], acc[ti][2], acc[ti][3]);
    float4 hi = make_float4(acc[ti][4], acc[ti][5], acc[ti][6], acc[ti][7]);
    *reinterpret_cast<float4*>(ctxh + (size_t)d * MPAD + m0)     = lo;
    *reinterpret_cast<float4*>(ctxh + (size_t)d * MPAD + m0 + 4) = hi;
  }
}

// ---------------------------------------------------------------------------
// Phase 3: q' (per-row stabilizer), denominator, out = (q' @ ctx) / den
// ---------------------------------------------------------------------------
__global__ void __launch_bounds__(256)
perf_output(const float* __restrict__ q, const float* __restrict__ proj,
            const float* __restrict__ kcum_in, const float* __restrict__ ctx_in,
            float* __restrict__ out) {
  extern __shared__ char smem[];
  __bf16* projL = reinterpret_cast<__bf16*>(smem);   // MPAD*DDIM
  __bf16* ctxT  = projL + MPAD * DDIM;               // DDIM*MPAD [d][m] bf16
  __bf16* qbL   = ctxT + DDIM * MPAD;                // 64*DDIM   [n][d] bf16
  __bf16* qpL   = qbL + 64 * DDIM;                   // 64*MPAD   [n][m] bf16 (q')
  float*  qdL   = reinterpret_cast<float*>(qpL + 64 * MPAD); // 64*MPAD f32
  float*  kcumL = qdL + 64 * MPAD;                   // MPAD
  float*  diagL = kcumL + MPAD;                      // 64
  float*  rmaxL = diagL + 64;                        // 64
  float*  denL  = rmaxL + 64;                        // 64

  const int tid = threadIdx.x, lane = tid & 31, wv = tid >> 5;
  const int l15 = lane & 15, kb = (lane >> 4) * 8;
  const int head = blockIdx.y;
  const int n0 = blockIdx.x * 64;
  const float* qh   = q + (size_t)head * NSEQ * DDIM;
  const float* ctxh = ctx_in + (size_t)head * DDIM * MPAD;

  stage_proj(proj, projL, tid);
  for (int idx4 = tid; idx4 < DDIM * MPAD / 4; idx4 += 256) {
    float4 f = *reinterpret_cast<const float4*>(ctxh + idx4 * 4);
    store_bf16x4(ctxT + idx4 * 4, f);               // already [d][m]
  }
  for (int idx4 = tid; idx4 < 64 * DDIM / 4; idx4 += 256) {
    float4 f = *reinterpret_cast<const float4*>(qh + (size_t)n0 * DDIM + idx4 * 4);
    store_bf16x4(qbL + idx4 * 4, f);
  }
  for (int idx = tid; idx < MPAD; idx += 256) kcumL[idx] = kcum_in[head * MPAD + idx];
  if (tid < 64) {                                    // diag from f32 global
    const float4* qr = reinterpret_cast<const float4*>(qh + (size_t)(n0 + tid) * DDIM);
    float s = 0.f;
#pragma unroll
    for (int j = 0; j < 16; ++j) {
      float4 f = qr[j];
      s += f.x * f.x + f.y * f.y + f.z * f.z + f.w * f.w;
    }
    diagL[tid] = DIAGC * s;
  }
  __syncthreads();

  // q_dash chunk: 4 n-tiles x 18 m-tiles -> qdL f32
  for (int t = wv; t < 72; t += 8) {
    int nt = t / 18, mt = t % 18;
    const __bf16* ab = qbL + (nt * 16 + l15) * DDIM;
    v16bf a0 = frag_bf16_lds(ab, kb);
    v16bf a1 = frag_bf16_lds(ab, kb + 32);
    const __bf16* pb = projL + (mt * 16 + l15) * DDIM;
    v16bf b0 = frag_bf16_lds(pb, kb);
    v16bf b1 = frag_bf16_lds(pb, kb + 32);
    v8f c = {};
    c = wmma_bf16(a0, b0, c);
    c = wmma_bf16(a1, b1, c);
    int col = mt * 16 + l15;
#pragma unroll
    for (int i = 0; i < 8; ++i) {
      int row = nt * 16 + i + ((lane >> 4) << 3);
      qdL[row * MPAD + col] = c[i];
    }
  }
  __syncthreads();
  if (tid < 64) {                                    // per-row stabilizer
    float m = NEGINF;
    for (int mcol = 0; mcol < MFEAT; ++mcol) m = fmaxf(m, qdL[tid * MPAD + mcol]);
    rmaxL[tid] = m;
  }
  __syncthreads();
  for (int idx4 = tid; idx4 < 64 * MPAD / 4; idx4 += 256) {  // transform to q'
    int r = idx4 / (MPAD / 4), mc0 = (idx4 - r * (MPAD / 4)) * 4;
    float4 dd = *reinterpret_cast<const float4*>(qdL + idx4 * 4);
    float bias = diagL[r] + rmaxL[r];
    float4 val;
    val.x = (mc0 + 0 < MFEAT) ? RATIO * (__expf(dd.x - bias) + EPSF) : 0.f;
    val.y = (mc0 + 1 < MFEAT) ? RATIO * (__expf(dd.y - bias) + EPSF) : 0.f;
    val.z = (mc0 + 2 < MFEAT) ? RATIO * (__expf(dd.z - bias) + EPSF) : 0.f;
    val.w = (mc0 + 3 < MFEAT) ? RATIO * (__expf(dd.w - bias) + EPSF) : 0.f;
    *reinterpret_cast<float4*>(qdL + idx4 * 4) = val;  // keep f32 for denom
    store_bf16x4(qpL + idx4 * 4, val);
  }
  __syncthreads();
  if (tid < 64) {                                    // den = q' . k_cumsum
    float s = 0.f;
    for (int mcol = 0; mcol < MFEAT; ++mcol) s += qdL[tid * MPAD + mcol] * kcumL[mcol];
    denL[tid] = s;
  }
  __syncthreads();

  // out = q' @ ctx  (16 tiles: 4 nt x 4 dt; K = 288 = 9 steps of 32)
#pragma unroll
  for (int ti = 0; ti < 2; ++ti) {
    int t = wv + ti * 8;
    int nt = t >> 2, dt = t & 3;
    v8f c = {};
    for (int ks = 0; ks < MPAD / 32; ++ks) {
      v16bf a = frag_bf16_lds(qpL + (nt * 16 + l15) * MPAD + ks * 32, kb);
      v16bf b = frag_bf16_lds(ctxT + (dt * 16 + l15) * MPAD + ks * 32, kb);
      c = wmma_bf16(a, b, c);
    }
#pragma unroll
    for (int i = 0; i < 8; ++i) {
      int row = nt * 16 + i + ((lane >> 4) << 3);
      out[((size_t)head * NSEQ + n0 + row) * DDIM + dt * 16 + l15] = c[i] / denL[row];
    }
  }
}

// ---------------------------------------------------------------------------
extern "C" void kernel_launch(void* const* d_in, const int* in_sizes, int n_in,
                              void* d_out, int out_size, void* d_ws, size_t ws_size,
                              hipStream_t stream) {
  (void)in_sizes; (void)n_in; (void)out_size; (void)ws_size;
  const float* q    = (const float*)d_in[0];
  const float* k    = (const float*)d_in[1];
  const float* v    = (const float*)d_in[2];
  const float* proj = (const float*)d_in[3];
  float* out = (float*)d_out;
  float* ws  = (float*)d_ws;

  float* kmax = ws;                          // 64
  float* kcum = ws + 64;                     // 64 * MPAD
  float* ctx  = kcum + (size_t)NHEAD * MPAD; // 64 * DDIM * MPAD (~4.7 MB), [d][m]

  size_t s1 = (size_t)MPAD * DDIM * 2 + 64;                              // ~36.9 KB
  size_t s2 = (size_t)(MPAD * DDIM + 32 * DDIM + DDIM * 32 + MPAD * 32) * 2 +
              (8 * MPAD + 32 + 4 * 32 * DDIM) * sizeof(float);           // ~105.6 KB
  size_t s3 = (size_t)(MPAD * DDIM + DDIM * MPAD + 64 * DDIM + 64 * MPAD) * 2 +
              ((size_t)64 * MPAD + MPAD + 3 * 64) * sizeof(float);       // ~193 KB

  perf_kmax   <<<NHEAD, 256, s1, stream>>>(k, proj, kmax);
  perf_context<<<NHEAD, 256, s2, stream>>>(k, v, proj, kmax, kcum, ctx);
  perf_output <<<dim3(NSEQ / 64, NHEAD), 256, s3, stream>>>(q, proj, kcum, ctx, out);
}
